// RelationalAttentionContextEncoder2_28845000360166
// MI455X (gfx1250) — matrix operand order repaired and checked by reading, not compiled, not run on hardware
//
#include <hip/hip_runtime.h>
#include <hip/hip_bf16.h>

typedef __attribute__((ext_vector_type(16))) _Float16 v16h;
typedef __attribute__((ext_vector_type(8)))  float    v8f;

#define NUM_ENT   7
#define DIM_ENT   4
#define HALFDIM   128
#define REL_IN    9
#define OUT_C     256                    // per-entity output channels (prop 128 | pooled 128)
#define ROW_STRIDE (NUM_ENT * OUT_C)     // 1792 floats per batch element

__device__ __forceinline__ float relu1(float x) {
  // v_med3_f32(x, 0, FLT_MAX): true clamp -> NOT foldable to maxnum, single VALU op.
  // (differs from max(x,0) only at x=+inf, which cannot occur here)
  return __builtin_amdgcn_fmed3f(x, 0.0f, 3.402823466e38f);
}

// One wave handles 16 batch rows (M=16) x 64 output channels (4 N-tiles).
// The wave id's low bit picks which N-half of the 128 rel channels it owns.
// Bias is folded into the GEMM as feature K=9 (A element = 1.0, B element =
// b_rel[h]), so the WMMA C operand is the inline constant 0 -> no splat movs.
__global__ __launch_bounds__(256, 4) void rel_pool_kernel(
    const float* __restrict__ ctx,    // (28, B) feature-major
    const float* __restrict__ W_rel,  // (128, 9)
    const float* __restrict__ b_rel,  // (128)
    float* __restrict__ out,          // (B, 7, 256), we write [...,128:256]
    int Btot)
{
  const int lane  = threadIdx.x & 31;
  const int wave  = threadIdx.x >> 5;
  const int gwave = blockIdx.x * 8 + wave;     // global wave id
  const int half  = gwave & 1;                 // N-half (tiles 0-3 or 4-7)
  const int b_base = (gwave >> 1) * 16;
  if (b_base >= Btot) return;                  // wave-uniform; EXEC stays all-ones
  const int n   = lane & 15;                   // N index in tile / batch row for loads
  const bool lo = lane < 16;
  const int b_row = b_base + n;                // both lane halves mirror rows 0..15

  // --- per-lane entity data (28 floats), reused for all 42 pairs ------------
  float ent[NUM_ENT * DIM_ENT];
#pragma unroll
  for (int f = 0; f < NUM_ENT * DIM_ENT; ++f)
    ent[f] = ctx[(size_t)f * (size_t)Btot + b_row];

  const _Float16 z16 = (_Float16)0.0f;
  const _Float16 one16 = (_Float16)1.0f;

  // --- B tiles (K=32 x N=16, f16), bias folded at K=9 -----------------------
  // B layout (table order): lanes 0-15 elems 0..7 = K0..7, 8..15 = K16..23; lanes 16-31 elems 0..7 = K8..15, 8..15 = K24..31.
  // K=0..8 = W_rel[h][0..8], K=9 = b_rel[h]; everything else zero padding.
  v16h Bt[4];
#pragma unroll
  for (int tt = 0; tt < 4; ++tt) {
    const int h = (half * 4 + tt) * 16 + n;
    const float* wr = W_rel + h * REL_IN;
    v16h bt = {};
#pragma unroll
    for (int k = 0; k < REL_IN - 1; ++k)
      bt[k] = lo ? (_Float16)wr[k] : z16;
    bt[0] = lo ? (_Float16)wr[0] : (_Float16)wr[8];
    bt[1] = lo ? (_Float16)wr[1] : (_Float16)b_rel[h];   // bias row (K=9)
    Bt[tt] = bt;
  }

  const v8f czero = {0.f, 0.f, 0.f, 0.f, 0.f, 0.f, 0.f, 0.f};
  const float inv6 = 1.0f / 6.0f;

#pragma unroll
  for (int e = 0; e < NUM_ENT; ++e) {
    v8f acc[4];
#pragma unroll
    for (int jj = 0; jj < 6; ++jj) {
      const int j = jj + (jj >= e ? 1 : 0);    // PJ: 0..6 skipping e
      const float d0 = ent[e * 4 + 0] - ent[j * 4 + 0];
      const float d1 = ent[e * 4 + 1] - ent[j * 4 + 1];
      const float d2 = ent[e * 4 + 2] - ent[j * 4 + 2];
      const float d3 = ent[e * 4 + 3] - ent[j * 4 + 3];
      const float dist = __builtin_amdgcn_sqrtf(d0 * d0 + d1 * d1); // raw v_sqrt_f32

      // A layout (16-bit A 16x32): lanes 0-15: elems 0..7 = K0..7 (features),
      // lanes 16-31: elem 0 = K8 (dist), elem 1 = K9 (1.0 bias feature);
      // all other K positions zero pad.
      v16h a = {};
      a[0] = lo ? (_Float16)ent[e * 4 + 2] : (_Float16)dist;
      a[1] = lo ? (_Float16)ent[e * 4 + 3] : one16;
      a[2] = lo ? (_Float16)ent[j * 4 + 2] : z16;
      a[3] = lo ? (_Float16)ent[j * 4 + 3] : z16;
      a[4] = lo ? (_Float16)d0 : z16;
      a[5] = lo ? (_Float16)d1 : z16;
      a[6] = lo ? (_Float16)d2 : z16;
      a[7] = lo ? (_Float16)d3 : z16;

      // ---- compute phase: 4 independent WMMAs (distinct D groups) ----------
      // Back-to-back WMMAs have no D->A/B hazard, so the 4-slot WMMA->VALU
      // RAW gap is filled with real work instead of v_nops.
      v8f dd[4];
#pragma unroll
      for (int tt = 0; tt < 4; ++tt)
        dd[tt] = __builtin_amdgcn_wmma_f32_16x16x32_f16(
            false, a, false, Bt[tt], (short)0, czero, false, false);

      // ---- accumulate phase: ReLU (v_med3) + pool-sum ----------------------
      if (jj == 0) {
#pragma unroll
        for (int tt = 0; tt < 4; ++tt)
#pragma unroll
          for (int r = 0; r < 8; ++r) acc[tt][r] = relu1(dd[tt][r]);
      } else {
#pragma unroll
        for (int tt = 0; tt < 4; ++tt)
#pragma unroll
          for (int r = 0; r < 8; ++r) acc[tt][r] += relu1(dd[tt][r]);
      }
    }
    // store pooled: out[b, e, 128 + h]
#pragma unroll
    for (int tt = 0; tt < 4; ++tt) {
      const int t = half * 4 + tt;
#pragma unroll
      for (int r = 0; r < 8; ++r) {
        const int M = r + (lo ? 0 : 8);        // C/D row mapping
        out[(size_t)(b_base + M) * ROW_STRIDE +
            (size_t)e * OUT_C + HALFDIM + t * 16 + n] = acc[tt][r] * inv6;
      }
    }
  }
}

// prop = relu(ents[...,2:] @ W_prop^T + b_prop) -> out[b, e, 0:128]
// Pure bandwidth job: 2 FMAs per output element, fully coalesced stores.
__global__ __launch_bounds__(256) void prop_kernel(
    const float* __restrict__ ctx, const float* __restrict__ W_prop,
    const float* __restrict__ b_prop, float* __restrict__ out, int Btot)
{
  const int tid = blockIdx.x * blockDim.x + threadIdx.x;
  const int h = tid & (HALFDIM - 1);
  const int e = (tid >> 7) % NUM_ENT;
  const int b = tid / (NUM_ENT * HALFDIM);
  if (b >= Btot) return;
  const float z = ctx[(size_t)(e * 4 + 2) * (size_t)Btot + b];
  const float w = ctx[(size_t)(e * 4 + 3) * (size_t)Btot + b];
  const float v = fmaf(z, W_prop[h * 2 + 0], fmaf(w, W_prop[h * 2 + 1], b_prop[h]));
  out[(size_t)b * ROW_STRIDE + (size_t)e * OUT_C + h] = relu1(v);
}

extern "C" void kernel_launch(void* const* d_in, const int* in_sizes, int n_in,
                              void* d_out, int out_size, void* d_ws, size_t ws_size,
                              hipStream_t stream) {
  const float* ctx    = (const float*)d_in[0];
  const float* W_rel  = (const float*)d_in[1];
  const float* b_rel  = (const float*)d_in[2];
  const float* W_prop = (const float*)d_in[3];
  const float* b_prop = (const float*)d_in[4];
  float* out = (float*)d_out;

  const int B = in_sizes[0] / (NUM_ENT * DIM_ENT);   // 32768

  // two waves (one per N-half) per 16 batch rows; 8 waves (256 thr) per block
  const int nwaves = ((B + 15) / 16) * 2;
  const int blocks = (nwaves + 7) / 8;
  rel_pool_kernel<<<blocks, 256, 0, stream>>>(ctx, W_rel, b_rel, out, B);

  const int total = B * NUM_ENT * HALFDIM;
  prop_kernel<<<(total + 255) / 256, 256, 0, stream>>>(ctx, W_prop, b_prop, out, B);
}
